// KAST_14396730376576
// MI455X (gfx1250) — matrix operand order repaired and checked by reading, not compiled, not run on hardware
//
#include <hip/hip_runtime.h>
#include <hip/hip_bf16.h>

// ---------------------------------------------------------------------------
// CDNA5 (gfx1250) implementation. wave32, WMMA 16x16x32 bf16,
// LDS-staged memory GEMM with async global->LDS double buffering.
// ---------------------------------------------------------------------------

typedef __attribute__((ext_vector_type(16))) __bf16 v16bf;
typedef __attribute__((ext_vector_type(8)))  __bf16 v8bf;
typedef __attribute__((ext_vector_type(8)))  float  v8f;
typedef __attribute__((ext_vector_type(4)))  int    v4i;

#define BS 2
#define T  9
#define H  64
#define W  64
#define HW 4096
#define CK 256
#define CV 3
#define MMEM 500
#define MPAD 512
#define SLOTS 512       // padded slot stride in corr buffer (max live = 509)
#define MEMWAVES 8      // waves per block in mem_top1

// ---------------------------------------------------------------------------
// async global->LDS staging support (guarded; falls back to reg+ds_store)
// ---------------------------------------------------------------------------
#if defined(__has_builtin)
#if __has_builtin(__builtin_amdgcn_global_load_async_to_lds_b128)
#define ASYNC_MODE 1
#else
#define ASYNC_MODE 0
#endif
#else
#define ASYNC_MODE 0
#endif

// copy 16 bytes global -> LDS (async when available)
static __device__ __forceinline__ void stage16(const unsigned short* src,
                                               unsigned short* dst_lds) {
#if ASYNC_MODE == 1
    // signature (from compiler diagnostic): (v4i addrspace(1)*, v4i addrspace(3)*, imm offset, imm cpol)
    __attribute__((address_space(1))) v4i* g =
        (__attribute__((address_space(1))) v4i*)(void*)src;
    __attribute__((address_space(3))) v4i* l =
        (__attribute__((address_space(3))) v4i*)(void*)dst_lds;
    __builtin_amdgcn_global_load_async_to_lds_b128(g, l, 0, 0);
#else
    uint4 t = *(const uint4*)src;
    *(uint4*)dst_lds = t;
#endif
}

template <int N>
static __device__ __forceinline__ void wait_async() {
#if ASYNC_MODE == 1
#if __has_builtin(__builtin_amdgcn_s_wait_asynccnt)
    __builtin_amdgcn_s_wait_asynccnt((short)N);
#else
    asm volatile("s_wait_asynccnt %0" ::"i"(N) : "memory");
#endif
#endif
}

// round-to-nearest-even float -> bf16 bits
static __device__ __forceinline__ unsigned short f2bf(float f) {
    unsigned int u = __float_as_uint(f);
    unsigned int r = (u + 0x7FFFu + ((u >> 16) & 1u)) >> 16;
    return (unsigned short)r;
}

// ---------------------------------------------------------------------------
// conversion kernels
// ---------------------------------------------------------------------------
__global__ void f2bf_kernel(const float* __restrict__ in,
                            unsigned short* __restrict__ out, int n) {
    int i = blockIdx.x * blockDim.x + threadIdx.x;
    if (i < n) out[i] = f2bf(in[i]);
}

__global__ void mk_pad_kernel(const float* __restrict__ in,
                              unsigned short* __restrict__ out) {
    int i = blockIdx.x * blockDim.x + threadIdx.x;
    if (i >= BS * MPAD * CK) return;
    int c = i & (CK - 1);
    int m = (i >> 8) & (MPAD - 1);
    int b = i >> 17;
    out[i] = (m < MMEM) ? f2bf(in[((size_t)b * MMEM + m) * CK + c]) : (unsigned short)0;
}

__global__ void init_prev_kernel(const float* __restrict__ v,
                                 float* __restrict__ prevv) {
    int i = blockIdx.x * blockDim.x + threadIdx.x;
    if (i >= BS * HW * CV) return;
    int b = i / (HW * CV);
    int rem = i - b * (HW * CV);
    prevv[((size_t)b * T) * HW * CV + rem] = v[((size_t)b * T) * HW * CV + rem];
}

// ---------------------------------------------------------------------------
// Banded correlation via WMMA (templated on dilation RATE).
// D[m,n] = dot(k_i[x0+m], k_prev[x0+n+beta]) = corr(x0+m, dx = beta+n-m)
// ---------------------------------------------------------------------------
template <int RATE>
__global__ __launch_bounds__(32) void corr_wmma_kernel(
    const unsigned short* __restrict__ kbf,   // (bs,T,4096,256) bf16
    float* __restrict__ corr,                 // (bs,4096,SLOTS)
    int ti, int tp, int slotBase)
{
    const int lane = threadIdx.x;
    const int half = lane >> 4;
    const int lm   = lane & 15;

    const int b     = blockIdx.z;
    const int x0    = blockIdx.y * 16;
    const int y     = blockIdx.x / 13;
    const int dyIdx = blockIdx.x % 13;
    const int dy    = (dyIdx - 6) * RATE;
    const int yb    = y + dy;
    const bool rowValid = (yb >= 0) && (yb < H);

    const unsigned short* ka =
        kbf + ((((size_t)b * T + ti) * HW) + (size_t)y * W + (x0 + lm)) * CK;
    v16bf A[8];
#pragma unroll
    for (int kc = 0; kc < 8; ++kc) {
        int c0 = kc * 32 + half * 16;
        v8bf lo = *(const v8bf*)(ka + c0);
        v8bf hi = *(const v8bf*)(ka + c0 + 8);
        A[kc] = __builtin_shufflevector(lo, hi, 0,1,2,3,4,5,6,7,8,9,10,11,12,13,14,15);
    }

    constexpr int nB = (RATE == 1) ? 2 : 3;
#pragma unroll
    for (int bi = 0; bi < nB; ++bi) {
        constexpr int b0 = (RATE == 1) ? -6 : -12;
        const int beta = b0 + 16 * bi;
        const int xb   = x0 + lm + beta;
        const bool colValid = rowValid && (xb >= 0) && (xb < W);

        v16bf B[8];
        if (colValid) {
            const unsigned short* kb =
                kbf + ((((size_t)b * T + tp) * HW) + (size_t)yb * W + xb) * CK;
#pragma unroll
            for (int kc = 0; kc < 8; ++kc) {
                int c0 = kc * 32 + half * 16;
                B[kc] = *(const v16bf*)(kb + c0);
            }
        } else {
#pragma unroll
            for (int kc = 0; kc < 8; ++kc) { v16bf z = {}; B[kc] = z; }
        }

        v8f acc = {};
#pragma unroll
        for (int kc = 0; kc < 8; ++kc) {
            acc = __builtin_amdgcn_wmma_f32_16x16x32_bf16(
                false, A[kc], false, B[kc], (short)0, acc, false, false);
        }

#pragma unroll
        for (int r = 0; r < 8; ++r) {
            const int m  = r + half * 8;
            const int dx = beta + lm - m;
            bool ok; int ix;
            if (RATE == 1) { ok = (dx >= -6 && dx <= 6); ix = dx + 6; }
            else           { ok = (dx >= -12 && dx <= 12 && ((dx & 1) == 0)); ix = (dx >> 1) + 6; }
            if (ok) {
                const int p = y * W + x0 + m;
                corr[(((size_t)b * HW) + p) * SLOTS + slotBase + dyIdx * 13 + ix] =
                    acc[r] * (1.0f / 256.0f);
            }
        }
    }
}

// ---------------------------------------------------------------------------
// Memory top-1: cooperative block (8 waves, 128 pixels). m_k chunks are
// staged into LDS (double-buffered, async global->LDS) and shared by all
// waves. Fused GEMM + row max/argmax; corr_memory == max inner product.
// ---------------------------------------------------------------------------
__global__ __launch_bounds__(32 * MEMWAVES) void mem_top1_kernel(
    const unsigned short* __restrict__ kbf,    // (bs,T,4096,256) bf16
    const unsigned short* __restrict__ mkbf,   // (bs,512,256) bf16
    const float* __restrict__ mv,              // (bs,500,3) f32
    float* __restrict__ corr,                  // (bs,4096,SLOTS)
    float* __restrict__ topv,                  // (bs,4096,2,3)
    int ti, int slotIdx, int memIdx)
{
    __shared__ unsigned short smk[2][16 * CK];   // 2 x 8KB chunk buffers

    const int tid  = threadIdx.x;
    const int wave = tid >> 5;
    const int lane = tid & 31;
    const int half = lane >> 4;
    const int lm   = lane & 15;
    const int b    = blockIdx.y;
    const int p0   = blockIdx.x * (16 * MEMWAVES) + wave * 16;

    // per-wave A tile (16 pixels x 256 ch)
    const unsigned short* ka =
        kbf + (((size_t)b * T + ti) * HW + p0 + lm) * CK;
    v16bf A[8];
#pragma unroll
    for (int kc = 0; kc < 8; ++kc) {
        int c0 = kc * 32 + half * 16;
        v8bf lo = *(const v8bf*)(ka + c0);
        v8bf hi = *(const v8bf*)(ka + c0 + 8);
        A[kc] = __builtin_shufflevector(lo, hi, 0,1,2,3,4,5,6,7,8,9,10,11,12,13,14,15);
    }

    const unsigned short* mkb = mkbf + (size_t)b * MPAD * CK;
    // stage one 16-row chunk: 8KB, 256 threads x 32B (two 16B transfers)
    auto stage = [&](int mc, int buf) {
        const unsigned short* src = mkb + (size_t)mc * 16 * CK + tid * 16;
        unsigned short* dst = &smk[buf][tid * 16];
        stage16(src, dst);
        stage16(src + 8, dst + 8);
    };

    float best[8];
    int   bestm[8];
#pragma unroll
    for (int r = 0; r < 8; ++r) { best[r] = -3.4e38f; bestm[r] = 0; }

    constexpr int NCH = MPAD / 16;   // 32 chunks
    stage(0, 0);
    wait_async<0>();
    __syncthreads();

    for (int mc = 0; mc < NCH; ++mc) {
        const int cur = mc & 1;
        if (mc + 1 < NCH) {
            stage(mc + 1, cur ^ 1);
            wait_async<2>();           // everything except the 2 newest done
        } else {
            wait_async<0>();
        }
        __syncthreads();               // chunk `cur` visible to all waves

        const unsigned short* brow = &smk[cur][lm * CK];
        v16bf B[8];
#pragma unroll
        for (int kc = 0; kc < 8; ++kc) {
            int c0 = kc * 32 + half * 16;
            B[kc] = *(const v16bf*)(brow + c0);    // LDS reads
        }
        v8f acc = {};
#pragma unroll
        for (int kc = 0; kc < 8; ++kc) {
            acc = __builtin_amdgcn_wmma_f32_16x16x32_bf16(
                false, A[kc], false, B[kc], (short)0, acc, false, false);
        }
        const int mcol = mc * 16 + lm;
        const bool mok = (mcol < MMEM);
#pragma unroll
        for (int r = 0; r < 8; ++r) {
            float c = acc[r];
            if (mok && c > best[r]) { best[r] = c; bestm[r] = mcol; }
        }
        __syncthreads();               // done reading before next overwrite
    }

    // reduce max/argmax across the 16 lanes holding the same matrix row
#pragma unroll
    for (int r = 0; r < 8; ++r) {
#pragma unroll
        for (int off = 8; off >= 1; off >>= 1) {
            float ob = __shfl_xor(best[r], off, 32);
            int   om = __shfl_xor(bestm[r], off, 32);
            if (ob > best[r] || (ob == best[r] && om < bestm[r])) {
                best[r] = ob; bestm[r] = om;
            }
        }
    }

    if (lm == 0) {   // lanes 0 (rows 0-7) and 16 (rows 8-15)
#pragma unroll
        for (int r = 0; r < 8; ++r) {
            const int row = r + half * 8;
            const int p   = p0 + row;
            corr[((size_t)b * HW + p) * SLOTS + slotIdx] = best[r];
            const float* mvp = mv + ((size_t)b * MMEM + bestm[r]) * CV;
            float* tv = topv + (((size_t)b * HW + p) * 2 + memIdx) * CV;
            tv[0] = mvp[0]; tv[1] = mvp[1]; tv[2] = mvp[2];
        }
    }
}

// ---------------------------------------------------------------------------
// Softmax over all slots + weighted sum (cv=3) + sequential state update.
// ---------------------------------------------------------------------------
__global__ void finalize_kernel(
    const float* __restrict__ corr,
    const float* __restrict__ topv,
    const float* __restrict__ vin,
    const unsigned char* __restrict__ mask,
    float* __restrict__ prevv,
    float* __restrict__ out,
    int i, int nlags, int nmem)
{
    int idx = blockIdx.x * blockDim.x + threadIdx.x;
    if (idx >= BS * HW) return;
    const int b = idx >> 12;
    const int p = idx & (HW - 1);
    const int y = p >> 6, x = p & 63;

    const float* crow = corr + ((size_t)b * HW + p) * SLOTS;
    const int S = nlags * 169 + nmem;

    float mx = -3.4e38f;
    for (int s = 0; s < S; ++s) mx = fmaxf(mx, crow[s]);

    float den = 0.f, n0 = 0.f, n1 = 0.f, n2 = 0.f;
    const int lags[3] = {1, 3, 5};
    for (int l = 0; l < nlags; ++l) {
        const int rate = (l == 0) ? 1 : 2;
        const int t = i - lags[l];
        const float* pv = prevv + (((size_t)b * T + t) * HW) * CV;
        for (int j = 0; j < 169; ++j) {
            float w = expf(crow[l * 169 + j] - mx);
            den += w;
            int dy = (j / 13 - 6) * rate;
            int dx = (j % 13 - 6) * rate;
            int yy = y + dy, xx = x + dx;
            if (yy >= 0 && yy < H && xx >= 0 && xx < W) {
                const float* vv = pv + ((size_t)yy * W + xx) * CV;
                n0 += w * vv[0]; n1 += w * vv[1]; n2 += w * vv[2];
            }
        }
    }
    for (int mi = 0; mi < nmem; ++mi) {
        float w = expf(crow[nlags * 169 + mi] - mx);
        den += w;
        const float* tv = topv + (((size_t)b * HW + p) * 2 + mi) * CV;
        n0 += w * tv[0]; n1 += w * tv[1]; n2 += w * tv[2];
    }
    const float inv = 1.0f / den;
    const float o0 = n0 * inv, o1 = n1 * inv, o2 = n2 * inv;

    float* op = out + (((size_t)b * (T - 1) + (i - 1)) * HW + p) * CV;
    op[0] = o0; op[1] = o1; op[2] = o2;

    const bool mk = mask[b * T + i] != 0;
    const float* vi = vin + (((size_t)b * T + i) * HW + p) * CV;
    float* pvo = prevv + (((size_t)b * T + i) * HW + p) * CV;
    pvo[0] = mk ? vi[0] : o0;
    pvo[1] = mk ? vi[1] : o1;
    pvo[2] = mk ? vi[2] : o2;
}

// ---------------------------------------------------------------------------
// host launcher
// ---------------------------------------------------------------------------
extern "C" void kernel_launch(void* const* d_in, const int* in_sizes, int n_in,
                              void* d_out, int out_size, void* d_ws, size_t ws_size,
                              hipStream_t stream) {
    const float* k   = (const float*)d_in[0];
    const float* v   = (const float*)d_in[1];
    const float* mk0 = (const float*)d_in[2];
    const float* mv0 = (const float*)d_in[3];
    const float* mk5 = (const float*)d_in[4];
    const float* mv5 = (const float*)d_in[5];
    const unsigned char* mask = (const unsigned char*)d_in[6];
    float* out = (float*)d_out;

    char* ws = (char*)d_ws;
    size_t off = 0;
    auto carve = [&](size_t bytes) -> void* {
        void* p = ws + off;
        off = (off + bytes + 255) & ~(size_t)255;
        return p;
    };
    unsigned short* kbf   = (unsigned short*)carve((size_t)BS * T * HW * CK * 2);
    unsigned short* mk0bf = (unsigned short*)carve((size_t)BS * MPAD * CK * 2);
    unsigned short* mk5bf = (unsigned short*)carve((size_t)BS * MPAD * CK * 2);
    float* corr  = (float*)carve((size_t)BS * HW * SLOTS * 4);
    float* topv  = (float*)carve((size_t)BS * HW * 2 * CV * 4);
    float* prevv = (float*)carve((size_t)BS * T * HW * CV * 4);

    {
        int nk = BS * T * HW * CK;
        f2bf_kernel<<<(nk + 255) / 256, 256, 0, stream>>>(k, kbf, nk);
        int nm = BS * MPAD * CK;
        mk_pad_kernel<<<(nm + 255) / 256, 256, 0, stream>>>(mk0, mk0bf);
        mk_pad_kernel<<<(nm + 255) / 256, 256, 0, stream>>>(mk5, mk5bf);
        int nv = BS * HW * CV;
        init_prev_kernel<<<(nv + 255) / 256, 256, 0, stream>>>(v, prevv);
    }

    for (int i = 1; i < T; ++i) {
        const int nlags = (i >= 5) ? 3 : (i >= 3) ? 2 : 1;
        const int nmem  = (i >= 6) ? 2 : 1;

        dim3 cg(H * 13, W / 16, BS);
        corr_wmma_kernel<1><<<cg, 32, 0, stream>>>(kbf, corr, i, i - 1, 0);
        if (i >= 3)
            corr_wmma_kernel<2><<<cg, 32, 0, stream>>>(kbf, corr, i, i - 3, 169);
        if (i >= 5)
            corr_wmma_kernel<2><<<cg, 32, 0, stream>>>(kbf, corr, i, i - 5, 338);

        dim3 mg(HW / (16 * MEMWAVES), BS);
        mem_top1_kernel<<<mg, 32 * MEMWAVES, 0, stream>>>(
            kbf, mk0bf, mv0, corr, topv, i, nlags * 169, 0);
        if (i >= 6)
            mem_top1_kernel<<<mg, 32 * MEMWAVES, 0, stream>>>(
                kbf, mk5bf, mv5, corr, topv, i, nlags * 169 + 1, 1);

        int np = BS * HW;
        finalize_kernel<<<(np + 255) / 256, 256, 0, stream>>>(
            corr, topv, v, mask, prevv, out, i, nlags, nmem);
    }
}
